// AwareDecoder_17025250361968
// MI455X (gfx1250) — matrix operand order repaired and checked by reading, not compiled, not run on hardware
//
#include <hip/hip_runtime.h>
#include <math.h>

// ---------------- constants from the reference ----------------
#define BB 64
#define SS 512
#define HH 768
#define LL 2
#define OPN 18
#define CC 24
#define MM 16
#define EE 8
#define AA 3
#define KK 43          // C + M + A
#define HD 21          // K // 2
#define H2 1536        // 2*H
#define H3 2304        // 3*H

typedef __attribute__((ext_vector_type(16))) __bf16 v16bf;
typedef __attribute__((ext_vector_type(8)))  __bf16 v8bf;
typedef __attribute__((ext_vector_type(8)))  float  v8f;

// ---------------- helpers ----------------
__device__ __forceinline__ __bf16 f2bf(float f) {
  union { float f; unsigned u; } v; v.f = f;
  unsigned r = v.u + 0x7fffu + ((v.u >> 16) & 1u);   // round-to-nearest-even
  unsigned short h = (unsigned short)(r >> 16);
  return __builtin_bit_cast(__bf16, h);
}

// A-tile (16x32 bf16) from pre-packed bf16 rows. Caller hands pointer at
// [row, k0+kba] (16B aligned). Elements 0-7 <- K=kba+0..7, 8-15 <- K=16+kba+0..7.
__device__ __forceinline__ v16bf load_a_bf16(const __bf16* p) {
  v8bf lo = *(const v8bf*)p;
  v8bf hi = *(const v8bf*)(p + 16);
  return __builtin_shufflevector(lo, hi, 0, 1, 2, 3, 4, 5, 6, 7,
                                         8, 9, 10, 11, 12, 13, 14, 15);
}

// B-tile (32x16): W stored row-major (N_out x K_in) bf16; elements 0-7 <- K=kb+0..7,
// 8-15 <- K=16+kb+0..7 (kb = 0 or 8), two aligned 16B vector loads.
__device__ __forceinline__ v16bf load_b_bf16(const __bf16* W, int ldw, int n0, int k0) {
  int lane = threadIdx.x & 31;
  int n = lane & 15;
  int kb = (lane < 16) ? 0 : 8;
  const __bf16* p = W + (size_t)(n0 + n) * ldw + k0 + kb;
  v8bf lo = *(const v8bf*)p;
  v8bf hi = *(const v8bf*)(p + 16);
  return __builtin_shufflevector(lo, hi, 0, 1, 2, 3, 4, 5, 6, 7,
                                         8, 9, 10, 11, 12, 13, 14, 15);
}

__device__ __forceinline__ const float* pick_src(int idx, int b,
    const float* pc, const float* pn, const float* prev) {
  if (idx >= 0 && idx < CC)                 return pc + (size_t)idx * HH;
  if (idx >= CC && idx < CC + MM)           return pn + ((size_t)b * MM + (idx - CC)) * HH;
  if (idx >= CC + MM && idx < CC + MM + EE) return prev + ((size_t)b * EE + (idx - CC - MM)) * HH;
  return nullptr;
}

// ---------------- simple utility kernels ----------------
__global__ void f32_to_bf16_kernel(const float* __restrict__ src, __bf16* __restrict__ dst, int n) {
  int i = blockIdx.x * blockDim.x + threadIdx.x;
  if (i < n) dst[i] = f2bf(src[i]);
}

__global__ void fill_zero_kernel(float* p, int n) {
  int i = blockIdx.x * blockDim.x + threadIdx.x;
  if (i < n) p[i] = 0.0f;
}

// copy rows, producing BOTH an f32 copy and a bf16 mirror
__global__ void copy_rows_dual_kernel(const float* __restrict__ src, long lds_,
                                      float* __restrict__ dstf, __bf16* __restrict__ dstb,
                                      long ldd, int rows, int cols) {
  long i = (long)blockIdx.x * blockDim.x + threadIdx.x;
  if (i >= (long)rows * cols) return;
  int r = (int)(i / cols), c = (int)(i % cols);
  float v = src[(size_t)r * lds_ + c];
  dstf[(size_t)r * ldd + c] = v;
  dstb[(size_t)r * ldd + c] = f2bf(v);
}

// broadcast cv into both GRU hidden-state buffers (f32 + bf16 mirrors), one launch
__global__ void broadcast_hx_kernel(const float* __restrict__ cv,
                                    float* __restrict__ h0f, __bf16* __restrict__ h0b,
                                    float* __restrict__ h1f, __bf16* __restrict__ h1b) {
  int i = blockIdx.x * blockDim.x + threadIdx.x;
  if (i >= BB * HH) return;
  float v = cv[i];
  __bf16 vb = f2bf(v);
  h0f[i] = v; h0b[i] = vb;
  h1f[i] = v; h1b[i] = vb;
}

// gather rows (optional) from f32 src into zero-padded bf16 staging
__global__ void pack_gather_bf16_kernel(const float* __restrict__ src,
                                        const int* __restrict__ gather, int nrows,
                                        __bf16* __restrict__ dst, int cols, int rows_pad) {
  long i = (long)blockIdx.x * blockDim.x + threadIdx.x;
  if (i >= (long)rows_pad * cols) return;
  int r = (int)(i / cols), c = (int)(i % cols);
  float v = 0.0f;
  if (r < nrows) {
    int sr = gather ? gather[r] : r;
    v = src[(size_t)sr * cols + c];
  }
  dst[i] = f2bf(v);
}

// ---------------- first/last number-token gather (writes bf16 staging) ----------------
__global__ __launch_bounds__(256) void find_numvec_kernel(
    const int* __restrict__ nmask, const float* __restrict__ inp, __bf16* __restrict__ numvec) {
  int bm = blockIdx.x;          // b*M + m
  int b = bm >> 4, m = bm & 15; // M == 16
  int tid = threadIdx.x;
  __shared__ int sf[256], sl[256];
  int f = 0x7fffffff, l = -1;
  const int* row = nmask + (size_t)b * SS;
  for (int s = tid; s < SS; s += 256)
    if (row[s] == m + 1) { f = min(f, s); l = max(l, s); }
  sf[tid] = f; sl[tid] = l; __syncthreads();
  for (int off = 128; off; off >>= 1) {
    if (tid < off) { sf[tid] = min(sf[tid], sf[tid + off]); sl[tid] = max(sl[tid], sl[tid + off]); }
    __syncthreads();
  }
  int first = sf[0], last = sl[0];
  bool exists = (last >= 0);
  __bf16* dst = numvec + (size_t)bm * H2;
  const float* fv = inp + ((size_t)b * SS + (exists ? first : 0)) * HH;
  const float* lv = inp + ((size_t)b * SS + (exists ? last  : 0)) * HH;
  for (int c = tid; c < HH; c += 256) {
    dst[c]      = f2bf(exists ? fv[c] : 0.0f);
    dst[HH + c] = f2bf(exists ? lv[c] : 0.0f);
  }
}

// ---------------- fused WMMA projection (2H -> H) + LayerNorm ----------------
// srcbf: bf16, zero-padded to a multiple of 16 rows.
// grid.x = ceil(nrows/16); block = 256 (8 waves). Wave w owns col tiles w*6..w*6+5.
__global__ __launch_bounds__(256) void proj_ln_kernel(
    const __bf16* __restrict__ srcbf, int nrows,
    const __bf16* __restrict__ W, const float* __restrict__ bias,
    const float* __restrict__ g, const float* __restrict__ be,
    float* __restrict__ outf, __bf16* __restrict__ outb) {
  __shared__ float smem[16 * HH];  // 48 KB
  int tid = threadIdx.x, wave = tid >> 5, lane = tid & 31;
  int row0 = blockIdx.x * 16;
  int m = lane & 15;
  int kba = (lane < 16) ? 0 : 8;
  const __bf16* arow = srcbf + (size_t)(row0 + m) * H2;

  v8f acc[6] = {};
  for (int k0 = 0; k0 < H2; k0 += 32) {
    v16bf a = load_a_bf16(arow + k0 + kba);
#pragma unroll
    for (int t = 0; t < 6; ++t) {
      int n0 = (wave * 6 + t) * 16;
      v16bf bm = load_b_bf16(W, H2, n0, k0);
      acc[t] = __builtin_amdgcn_wmma_f32_16x16x32_bf16(false, a, false, bm,
                                                       (short)0, acc[t], false, false);
    }
  }
  int coll = lane & 15, rh = (lane >> 4) * 8;
#pragma unroll
  for (int t = 0; t < 6; ++t) {
    int col = (wave * 6 + t) * 16 + coll;
#pragma unroll
    for (int v = 0; v < 8; ++v) smem[(rh + v) * HH + col] = acc[t][v] + bias[col];
  }
  __syncthreads();
  // LayerNorm: wave handles rows 2w, 2w+1
  for (int rr = 0; rr < 2; ++rr) {
    int r = wave * 2 + rr;
    if (row0 + r >= nrows) continue;
    float s = 0.0f, s2 = 0.0f;
    for (int c = lane; c < HH; c += 32) { float v = smem[r * HH + c]; s += v; s2 += v * v; }
#pragma unroll
    for (int off = 16; off; off >>= 1) { s += __shfl_xor(s, off, 32); s2 += __shfl_xor(s2, off, 32); }
    float mean = s * (1.0f / HH);
    float var  = s2 * (1.0f / HH) - mean * mean;
    float inv  = rsqrtf(var + 1e-5f);
    for (int c = lane; c < HH; c += 32) {
      float v = (smem[r * HH + c] - mean) * inv * g[c] + be[c];
      if (outf) outf[(size_t)(row0 + r) * HH + c] = v;
      if (outb) outb[(size_t)(row0 + r) * HH + c] = f2bf(v);
    }
  }
}

// ---------------- one GRU layer (fused WMMA GEMMs + gates) ----------------
// grid.x = 48 (col tiles of 16 over H=768); block = 128 (4 waves = 4 row tiles of 16).
__global__ __launch_bounds__(128) void gru_layer_kernel(
    const __bf16* __restrict__ xbf, long ldx,   // B x H bf16 A-matrix
    const __bf16* __restrict__ hbf,             // B x H bf16 A-matrix (stride ldh)
    const float* __restrict__ hprev, long ldh,  // f32 view for the gate term
    const __bf16* __restrict__ Wih,             // 3H x H bf16
    const __bf16* __restrict__ Whh,             // 3H x H bf16
    const float* __restrict__ bih, const float* __restrict__ bhh,
    float* __restrict__ hout, __bf16* __restrict__ hout_bf) {  // B x H, ld = H
  int wave = threadIdx.x >> 5, lane = threadIdx.x & 31;
  int n0 = blockIdx.x * 16;
  int row0 = wave * 16;
  int m = lane & 15;
  int kba = (lane < 16) ? 0 : 8;
  const __bf16* xrow = xbf + (size_t)(row0 + m) * ldx;
  const __bf16* hrow = hbf + (size_t)(row0 + m) * ldh;

  v8f air = {}, aiz = {}, ain = {}, ahr = {}, ahz = {}, ahn = {};
  for (int k0 = 0; k0 < HH; k0 += 32) {
    v16bf ax = load_a_bf16(xrow + k0 + kba);
    v16bf ah = load_a_bf16(hrow + k0 + kba);
    v16bf br = load_b_bf16(Wih, HH, n0,          k0);
    v16bf bz = load_b_bf16(Wih, HH, n0 + HH,     k0);
    v16bf bn = load_b_bf16(Wih, HH, n0 + 2 * HH, k0);
    v16bf cr = load_b_bf16(Whh, HH, n0,          k0);
    v16bf cz = load_b_bf16(Whh, HH, n0 + HH,     k0);
    v16bf cn = load_b_bf16(Whh, HH, n0 + 2 * HH, k0);
    air = __builtin_amdgcn_wmma_f32_16x16x32_bf16(false, ax, false, br, (short)0, air, false, false);
    aiz = __builtin_amdgcn_wmma_f32_16x16x32_bf16(false, ax, false, bz, (short)0, aiz, false, false);
    ain = __builtin_amdgcn_wmma_f32_16x16x32_bf16(false, ax, false, bn, (short)0, ain, false, false);
    ahr = __builtin_amdgcn_wmma_f32_16x16x32_bf16(false, ah, false, cr, (short)0, ahr, false, false);
    ahz = __builtin_amdgcn_wmma_f32_16x16x32_bf16(false, ah, false, cz, (short)0, ahz, false, false);
    ahn = __builtin_amdgcn_wmma_f32_16x16x32_bf16(false, ah, false, cn, (short)0, ahn, false, false);
  }
  // gates from C layout: VGPR v, lane -> row = v + 8*(lane>=16), col = lane&15
  int col = n0 + (lane & 15);
  int rh = (lane >> 4) * 8;
  float b_ir = bih[col],           b_hr = bhh[col];
  float b_iz = bih[HH + col],      b_hz = bhh[HH + col];
  float b_in = bih[2 * HH + col],  b_hn = bhh[2 * HH + col];
#pragma unroll
  for (int v = 0; v < 8; ++v) {
    int row = row0 + rh + v;
    float ir = air[v] + b_ir, hr = ahr[v] + b_hr;
    float iz = aiz[v] + b_iz, hz = ahz[v] + b_hz;
    float in_ = ain[v] + b_in, hn = ahn[v] + b_hn;
    float r = 1.0f / (1.0f + __expf(-(ir + hr)));
    float z = 1.0f / (1.0f + __expf(-(iz + hz)));
    float n = tanhf(in_ + r * hn);
    float hp = hprev[(size_t)row * ldh + col];
    float h = (1.0f - z) * n + z * hp;
    hout[(size_t)row * HH + col] = h;
    hout_bf[(size_t)row * HH + col] = f2bf(h);
  }
}

// ---------------- fused operator head: logits = W2·relu(W1·cv + b1) + b2 ----------------
// grid = BB blocks, 256 threads (8 waves). Hidden dim 384.
__global__ __launch_bounds__(256) void opc_head_kernel(
    const float* __restrict__ x,
    const float* __restrict__ W1, const float* __restrict__ b1,
    const float* __restrict__ W2, const float* __restrict__ b2,
    float* __restrict__ out, int ldo) {
  __shared__ float sh[384];
  int b = blockIdx.x, tid = threadIdx.x, wave = tid >> 5, lane = tid & 31;
  const float* xr = x + (size_t)b * HH;
  for (int o = wave; o < 384; o += 8) {
    const float* w = W1 + (size_t)o * HH;
    float s = 0.0f;
    for (int c = lane; c < HH; c += 32) s += xr[c] * w[c];
#pragma unroll
    for (int off = 16; off; off >>= 1) s += __shfl_xor(s, off, 32);
    if (lane == 0) sh[o] = fmaxf(s + b1[o], 0.0f);
  }
  __syncthreads();
  for (int o = tid; o < OPN; o += 256) {
    const float* w = W2 + (size_t)o * 384;
    float s = b2[o];
    for (int k = 0; k < 384; ++k) s += sh[k] * w[k];
    out[(size_t)b * ldo + o] = s;
  }
}

// ---------------- fused operand head + argmax + prev update + pred pick ----------------
// grid = BB blocks, 256 threads. Does: h=relu(W1·x+b1); logits=W2·h+b2 (written to
// d_out); oidx=argmax; empty=max(prev[b,i])==0 (pre-update); cond update; pred=pick.
__global__ __launch_bounds__(256) void odc_post_kernel(
    const float* __restrict__ x,
    const float* __restrict__ W1, const float* __restrict__ b1,
    const float* __restrict__ W2, const float* __restrict__ b2,
    float* __restrict__ outlogits, int ldo,
    float* __restrict__ prev, int i,
    const float* __restrict__ pc, const float* __restrict__ pn,
    float* __restrict__ pred) {
  __shared__ float sh[HD];
  __shared__ float sl[KK];
  __shared__ float sred[256];
  __shared__ int s_oidx, s_cond;
  int b = blockIdx.x, tid = threadIdx.x, wave = tid >> 5, lane = tid & 31;
  const float* xr = x + (size_t)b * HH;
  float* pr = prev + ((size_t)b * EE + i) * HH;

  // pre-update emptiness reduction (partial per-thread)
  float mx = -3.4e38f;
  for (int c = tid; c < HH; c += 256) mx = fmaxf(mx, pr[c]);
  sred[tid] = mx;

  // hidden layer (21 outputs, one per wave round-robin)
  for (int o = wave; o < HD; o += 8) {
    const float* w = W1 + (size_t)o * HH;
    float s = 0.0f;
    for (int c = lane; c < HH; c += 32) s += xr[c] * w[c];
#pragma unroll
    for (int off = 16; off; off >>= 1) s += __shfl_xor(s, off, 32);
    if (lane == 0) sh[o] = fmaxf(s + b1[o], 0.0f);
  }
  __syncthreads();
  for (int off = 128; off; off >>= 1) {
    if (tid < off) sred[tid] = fmaxf(sred[tid], sred[tid + off]);
    __syncthreads();
  }
  // output logits
  if (tid < KK) {
    const float* w = W2 + (size_t)tid * HD;
    float s = b2[tid];
    for (int k = 0; k < HD; ++k) s += sh[k] * w[k];
    sl[tid] = s;
    outlogits[(size_t)b * ldo + tid] = s;
  }
  __syncthreads();
  if (tid == 0) {
    int best = 0; float bv = sl[0];
    for (int k = 1; k < KK; ++k) if (sl[k] > bv) { bv = sl[k]; best = k; }
    s_oidx = best;
    s_cond = (best == 0) && (sred[0] == 0.0f);   // PAD==0 implies < C
  }
  __syncthreads();
  if (s_cond) for (int c = tid; c < HH; c += 256) pr[c] = xr[c];
  __syncthreads();
  const float* src = pick_src(s_oidx, b, pc, pn, prev);
  for (int c = tid; c < HH; c += 256) pred[(size_t)b * HH + c] = src ? src[c] : 0.0f;
}

// ---------------- decode bookkeeping kernels ----------------
__global__ void pick_kernel(const int* __restrict__ idxp, int idx_stride,
                            const float* __restrict__ pc, const float* __restrict__ pn,
                            const float* __restrict__ prev, __bf16* __restrict__ xout_bf) {
  int b = blockIdx.x, tid = threadIdx.x;
  int idx = idxp[(size_t)b * idx_stride];
  const float* src = pick_src(idx, b, pc, pn, prev);
  __bf16* dst = xout_bf + (size_t)b * HH;
  for (int c = tid; c < HH; c += 256) dst[c] = f2bf(src ? src[c] : 0.0f);
}

// finalize prev slice i, then refresh its bf16 mirror (used as next ctx A-matrix)
__global__ __launch_bounds__(256) void finalize_kernel(float* __restrict__ prev, int i,
                                                       const float* __restrict__ pred,
                                                       __bf16* __restrict__ prev_bf) {
  int b = blockIdx.x, tid = threadIdx.x;
  __shared__ float sred[256];
  float* pr = prev + ((size_t)b * EE + i) * HH;
  __bf16* prb = prev_bf + ((size_t)b * EE + i) * HH;
  float mx = -3.4e38f;
  for (int c = tid; c < HH; c += 256) mx = fmaxf(mx, pr[c]);
  sred[tid] = mx; __syncthreads();
  for (int off = 128; off; off >>= 1) {
    if (tid < off) sred[tid] = fmaxf(sred[tid], sred[tid + off]);
    __syncthreads();
  }
  if (sred[0] == 0.0f)
    for (int c = tid; c < HH; c += 256) pr[c] = pred[(size_t)b * HH + c];
  __syncthreads();
  for (int c = tid; c < HH; c += 256) prb[c] = f2bf(pr[c]);
}

// ---------------- host orchestration ----------------
extern "C" void kernel_launch(void* const* d_in, const int* in_sizes, int n_in,
                              void* d_out, int out_size, void* d_ws, size_t ws_size,
                              hipStream_t stream) {
  (void)in_sizes; (void)n_in; (void)out_size; (void)ws_size;
  const float* input           = (const float*)d_in[0];
  const int*   number_mask     = (const int*)d_in[3];
  const int*   gold_operators  = (const int*)d_in[4];
  const int*   gold_operands   = (const int*)d_in[5];
  const float* const_vector    = (const float*)d_in[6];
  const float* operator_vector = (const float*)d_in[7];
  const float* opj_W = (const float*)d_in[8];
  const float* opj_b = (const float*)d_in[9];
  const float* opj_g = (const float*)d_in[10];
  const float* opj_be = (const float*)d_in[11];
  const float* odj_W = (const float*)d_in[12];
  const float* odj_b = (const float*)d_in[13];
  const float* odj_g = (const float*)d_in[14];
  const float* odj_be = (const float*)d_in[15];
  const float* opc_W1 = (const float*)d_in[16];
  const float* opc_b1 = (const float*)d_in[17];
  const float* opc_W2 = (const float*)d_in[18];
  const float* opc_b2 = (const float*)d_in[19];
  const float* odc_W1 = (const float*)d_in[20];
  const float* odc_b1 = (const float*)d_in[21];
  const float* odc_W2 = (const float*)d_in[22];
  const float* odc_b2 = (const float*)d_in[23];
  const float* og_Wih = (const float*)d_in[24];
  const float* og_Whh = (const float*)d_in[25];
  const float* og_bih = (const float*)d_in[26];
  const float* og_bhh = (const float*)d_in[27];
  const float* cg_Wih = (const float*)d_in[28];
  const float* cg_Whh = (const float*)d_in[29];
  const float* cg_bih = (const float*)d_in[30];
  const float* cg_bhh = (const float*)d_in[31];

  float* out = (float*)d_out;

  // -------- workspace carve --------
  char* base = (char*)d_ws;
  size_t off = 0;
  auto alloc = [&](size_t bytes) -> void* {
    void* p = base + off; off = (off + bytes + 255) & ~(size_t)255; return p;
  };
  const size_t NGRU = (size_t)LL * H3 * HH;   // 2*2304*768
  const size_t NPRJ = (size_t)HH * H2;        // 768*1536
  __bf16* og_Wih_bf = (__bf16*)alloc(NGRU * 2);
  __bf16* og_Whh_bf = (__bf16*)alloc(NGRU * 2);
  __bf16* cg_Wih_bf = (__bf16*)alloc(NGRU * 2);
  __bf16* cg_Whh_bf = (__bf16*)alloc(NGRU * 2);
  __bf16* odj_W_bf  = (__bf16*)alloc(NPRJ * 2);
  __bf16* opj_W_bf  = (__bf16*)alloc(NPRJ * 2);
  // bf16 activation stagings
  __bf16* nv_bf     = (__bf16*)alloc((size_t)BB * MM * H2 * 2);  // num_vec
  __bf16* pc_src_bf = (__bf16*)alloc((size_t)32 * H2 * 2);       // const_vector padded 24->32
  __bf16* gv_src_bf = (__bf16*)alloc((size_t)BB * EE * H2 * 2);  // gathered operator_vector
  __bf16* gopv_bf   = (__bf16*)alloc((size_t)BB * EE * HH * 2);  // projected gold_opv (bf16 only)
  __bf16* prev_bf   = (__bf16*)alloc((size_t)BB * EE * HH * 2);
  __bf16* cv_bf[2]  = { (__bf16*)alloc((size_t)BB * HH * 2), (__bf16*)alloc((size_t)BB * HH * 2) };
  __bf16* t0_bf     = (__bf16*)alloc((size_t)BB * HH * 2);
  __bf16* hx0_bf[2] = { (__bf16*)alloc((size_t)BB * HH * 2), (__bf16*)alloc((size_t)BB * HH * 2) };
  __bf16* hx1_bf[2] = { (__bf16*)alloc((size_t)BB * HH * 2), (__bf16*)alloc((size_t)BB * HH * 2) };
  __bf16* xbuf_bf   = (__bf16*)alloc((size_t)BB * HH * 2);
  // f32 buffers
  float* pn     = (float*)alloc((size_t)BB * MM * HH * 4);
  float* pcb    = (float*)alloc((size_t)CC * HH * 4);
  float* prev   = (float*)alloc((size_t)BB * EE * HH * 4);
  float* cv[2]  = { (float*)alloc((size_t)BB * HH * 4), (float*)alloc((size_t)BB * HH * 4) };
  float* t0     = (float*)alloc((size_t)BB * HH * 4);
  float* hx0[2] = { (float*)alloc((size_t)BB * HH * 4), (float*)alloc((size_t)BB * HH * 4) };
  float* hx1[2] = { (float*)alloc((size_t)BB * HH * 4), (float*)alloc((size_t)BB * HH * 4) };
  float* pred   = (float*)alloc((size_t)BB * HH * 4);

  // -------- weight conversion --------
  {
    int n = (int)NGRU;
    f32_to_bf16_kernel<<<(n + 255) / 256, 256, 0, stream>>>(og_Wih, og_Wih_bf, n);
    f32_to_bf16_kernel<<<(n + 255) / 256, 256, 0, stream>>>(og_Whh, og_Whh_bf, n);
    f32_to_bf16_kernel<<<(n + 255) / 256, 256, 0, stream>>>(cg_Wih, cg_Wih_bf, n);
    f32_to_bf16_kernel<<<(n + 255) / 256, 256, 0, stream>>>(cg_Whh, cg_Whh_bf, n);
    int p = (int)NPRJ;
    f32_to_bf16_kernel<<<(p + 255) / 256, 256, 0, stream>>>(odj_W, odj_W_bf, p);
    f32_to_bf16_kernel<<<(p + 255) / 256, 256, 0, stream>>>(opj_W, opj_W_bf, p);
  }

  // -------- build bf16 projection inputs --------
  find_numvec_kernel<<<BB * MM, 256, 0, stream>>>(number_mask, input, nv_bf);
  pack_gather_bf16_kernel<<<(32 * H2 + 255) / 256, 256, 0, stream>>>(
      const_vector, nullptr, CC, pc_src_bf, H2, 32);
  pack_gather_bf16_kernel<<<(BB * EE * H2 + 255) / 256, 256, 0, stream>>>(
      operator_vector, gold_operators, BB * EE, gv_src_bf, H2, BB * EE);

  // -------- projections (WMMA + LN) --------
  proj_ln_kernel<<<(BB * MM) / 16, 256, 0, stream>>>(nv_bf, BB * MM,
      odj_W_bf, odj_b, odj_g, odj_be, pn, nullptr);
  proj_ln_kernel<<<2, 256, 0, stream>>>(pc_src_bf, CC,
      odj_W_bf, odj_b, odj_g, odj_be, pcb, nullptr);
  proj_ln_kernel<<<(BB * EE) / 16, 256, 0, stream>>>(gv_src_bf, BB * EE,
      opj_W_bf, opj_b, opj_g, opj_be, nullptr, gopv_bf);

  // -------- init --------
  fill_zero_kernel<<<(BB * EE * HH + 255) / 256, 256, 0, stream>>>(prev, BB * EE * HH);
  copy_rows_dual_kernel<<<(BB * HH + 255) / 256, 256, 0, stream>>>(
      input, (long)SS * HH, cv[0], cv_bf[0], HH, BB, HH);
  int cvi = 0;
  const size_t GL = (size_t)H3 * HH;  // per-layer GRU weight stride

  // -------- decode loop --------
  for (int i = 0; i < EE; ++i) {
    if (i > 0) {
      const float*  ctx   = prev    + (size_t)(i - 1) * HH;  // row stride E*H
      const __bf16* ctxbf = prev_bf + (size_t)(i - 1) * HH;
      gru_layer_kernel<<<48, 128, 0, stream>>>(cv_bf[cvi], (long)HH, ctxbf, ctx, (long)EE * HH,
          cg_Wih_bf, cg_Whh_bf, cg_bih, cg_bhh, t0, t0_bf);
      gru_layer_kernel<<<48, 128, 0, stream>>>(t0_bf, (long)HH, ctxbf, ctx, (long)EE * HH,
          cg_Wih_bf + GL, cg_Whh_bf + GL, cg_bih + H3, cg_bhh + H3, cv[1 - cvi], cv_bf[1 - cvi]);
      cvi = 1 - cvi;
    }
    // op logits -> d_out[ b*E*OP + i*OP + o ]  (single fused launch)
    opc_head_kernel<<<BB, 256, 0, stream>>>(cv[cvi], opc_W1, opc_b1, opc_W2, opc_b2,
        out + (size_t)i * OPN, EE * OPN);

    // hx = broadcast cv (f32 + bf16 mirrors for both layers, one launch)
    broadcast_hx_kernel<<<(BB * HH + 255) / 256, 256, 0, stream>>>(
        cv[cvi], hx0[0], hx0_bf[0], hx1[0], hx1_bf[0]);
    int h0i = 0, h1i = 0;

    for (int j = 0; j < AA; ++j) {
      const __bf16* xin;
      long ldxin;
      if (j == 0) {
        xin = gopv_bf + (size_t)i * HH;   // row (b,i) at (b*E+i)*H
        ldxin = (long)EE * HH;
      } else {
        pick_kernel<<<BB, 256, 0, stream>>>(gold_operands + i * AA + (j - 1), EE * AA,
            pcb, pn, prev, xbuf_bf);
        xin = xbuf_bf;
        ldxin = (long)HH;
      }

      gru_layer_kernel<<<48, 128, 0, stream>>>(xin, ldxin, hx0_bf[h0i], hx0[h0i], (long)HH,
          og_Wih_bf, og_Whh_bf, og_bih, og_bhh, hx0[1 - h0i], hx0_bf[1 - h0i]);
      h0i = 1 - h0i;
      gru_layer_kernel<<<48, 128, 0, stream>>>(hx0_bf[h0i], (long)HH, hx1_bf[h1i], hx1[h1i], (long)HH,
          og_Wih_bf + GL, og_Whh_bf + GL, og_bih + H3, og_bhh + H3, hx1[1 - h1i], hx1_bf[1 - h1i]);
      h1i = 1 - h1i;
      const float* xcur = hx1[h1i];

      // od logits + argmax + prev update + pred pick, single fused launch
      float* odl = out + (size_t)BB * EE * OPN + (size_t)(i * AA + j) * KK;
      odc_post_kernel<<<BB, 256, 0, stream>>>(xcur, odc_W1, odc_b1, odc_W2, odc_b2,
          odl, EE * AA * KK, prev, i, pcb, pn, pred);
    }
    finalize_kernel<<<BB, 256, 0, stream>>>(prev, i, pred, prev_bf);
  }
}
